// SwinLSTMCell_17617955848210
// MI455X (gfx1250) — matrix operand order repaired and checked by reading, not compiled, not run on hardware
//
#include <hip/hip_runtime.h>
#include <hip/hip_bf16.h>

typedef _Float16 v16h __attribute__((ext_vector_type(16)));
typedef float    v8f  __attribute__((ext_vector_type(8)));

#define HW   16384
#define WIM  128
#define CH   256
#define MTOT 65536
#define LDA  40   // 80-byte row stride: 16B-aligned fragment runs, conflict-free (gcd trick)

// ---- WMMA fragment helpers (wave32; A and B both use contiguous k = 16*hi + i) ----

// A-matrix 16x32 f16 from row-major LDS tile (row0 points at [m=0][k=0], ld in elems)
__device__ __forceinline__ v16h ld_afrag(const _Float16* row0, int ld, int lane) {
  int hi = (lane >> 4) & 1;
  int m  = lane & 15;
  const _Float16* r = row0 + m * ld;
  v16h a;
#pragma unroll
  for (int i = 0; i < 16; ++i) {
    int k = 16 * hi + i;
    a[i] = r[k];
  }
  return a;
}

// B-matrix 32x16 f16 from pre-converted f16 weights W[N][K] row-major (B[k][n]=W[n][k]).
// Per lane: 16 contiguous halves (32B) -> two global_load_b128.
__device__ __forceinline__ v16h ld_bfrag_h(const _Float16* W, int ldw, int n0, int k0, int lane) {
  int n = n0 + (lane & 15);
  const _Float16* p = W + (long)n * ldw + k0 + 16 * ((lane >> 4) & 1);
  v16h b;
#pragma unroll
  for (int i = 0; i < 16; ++i) b[i] = p[i];
  return b;
}

__device__ __forceinline__ float sigf(float x) { return 1.f / (1.f + __expf(-x)); }

// ---- fp32 -> f16 weight pre-conversion ----
__global__ void k_cvt(const float* __restrict__ src, _Float16* __restrict__ dst, int n) {
  int i = blockIdx.x * 256 + threadIdx.x;
  if (i < n) dst[i] = (_Float16)src[i];
}

// ---- Generic tiled WMMA GEMM: out[M][256] = A[M][K] * W[N][K]^T + bias ----
// MODE 0: A gathered from channel-major x (64 ch) ++ h (256 ch)  (in_proj)
// MODE 1: A = GroupNorm(projout) fused on load                    (gates)
template <int MODE>
__global__ __launch_bounds__(256) void k_gemm(
    const float* __restrict__ A0, const float* __restrict__ A1,
    const _Float16* __restrict__ W, const float* __restrict__ bias,
    const float* __restrict__ mu, const float* __restrict__ rsg,
    const float* __restrict__ gng, const float* __restrict__ gnb,
    float* __restrict__ out, int K) {
  __shared__ _Float16 sA[64][LDA];
  int t = threadIdx.x, w = t >> 5, lane = t & 31;
  int m0 = blockIdx.x * 64;
  int nb = blockIdx.y * 64;
  int mw = (w & 3) * 16;
  int nw = nb + (w >> 2) * 32;
  v8f acc0 = {}, acc1 = {};
  for (int kc = 0; kc < K; kc += 32) {
    if (MODE == 0) {
      int r = t & 63, kg = t >> 6;
      int m = m0 + r, b = m >> 14, hw = m & (HW - 1);
#pragma unroll
      for (int j = 0; j < 8; ++j) {
        int kl = kg * 8 + j, k = kc + kl;
        float v = (k < 64) ? A0[((long)b * 64 + k) * HW + hw]
                           : A1[((long)b * CH + (k - 64)) * HW + hw];
        sA[r][kl] = (_Float16)v;
      }
    } else {
      int kl = t & 31, rr = t >> 5;
      int c = kc + kl, g = c >> 5;
#pragma unroll
      for (int j = 0; j < 8; ++j) {
        int r = rr * 8 + j;
        int m = m0 + r, b = m >> 14;
        float v = A0[(long)m * CH + c];
        v = (v - mu[b * 8 + g]) * rsg[b * 8 + g] * gng[c] + gnb[c];
        sA[r][kl] = (_Float16)v;
      }
    }
    __syncthreads();
    v16h a  = ld_afrag(&sA[mw][0], LDA, lane);
    v16h b0 = ld_bfrag_h(W, K, nw, kc, lane);
    v16h b1 = ld_bfrag_h(W, K, nw + 16, kc, lane);
    acc0 = __builtin_amdgcn_wmma_f32_16x16x32_f16(false, a, false, b0, (short)0, acc0, false, false);
    acc1 = __builtin_amdgcn_wmma_f32_16x16x32_f16(false, a, false, b1, (short)0, acc1, false, false);
    __syncthreads();
  }
  int hi = (lane >> 4) & 1, nlo = lane & 15;
  float bs0 = bias[nw + nlo], bs1 = bias[nw + 16 + nlo];
#pragma unroll
  for (int j = 0; j < 8; ++j) {
    long m = m0 + mw + j + 8 * hi;
    out[m * CH + nw + nlo]      = acc0[j] + bs0;
    out[m * CH + nw + 16 + nlo] = acc1[j] + bs1;
  }
}

// ---- Fused per-window kernel: LayerNorm -> QKV GEMM -> attention -> proj ----
// One block per 4x4 window; 8 waves; wave w = head w for attention.
__global__ __launch_bounds__(256) void k_attn(
    const float* __restrict__ p, const float* __restrict__ lng, const float* __restrict__ lnb,
    const _Float16* __restrict__ qkvw, const float* __restrict__ qkvb,
    const _Float16* __restrict__ projw, const float* __restrict__ projb,
    float* __restrict__ outp) {
  __shared__ _Float16 sXN[16][CH];     // layer-normed tokens
  __shared__ _Float16 sQKV[16][768];   // q|k|v, head-major within each 256
  __shared__ _Float16 sP[8][16][18];   // per-wave softmax tiles (padded)
  __shared__ _Float16 sAO[16][CH];     // attention output
  int t = threadIdx.x, w = t >> 5, lane = t & 31;
  int wi = blockIdx.x;
  int b = wi >> 10, rem = wi & 1023, hb = rem >> 5, wb = rem & 31;
  int hi = (lane >> 4) & 1, nlo = lane & 15;

  // Phase 1: LayerNorm, 2 tokens per wave
#pragma unroll
  for (int s2 = 0; s2 < 2; ++s2) {
    int tok = 2 * w + s2;
    int row = hb * 4 + (tok >> 2), col = wb * 4 + (tok & 3);
    long m = (long)b * HW + row * WIM + col;
    const float* pr = p + m * CH;
    float vals[8], sum = 0.f, sq = 0.f;
#pragma unroll
    for (int j = 0; j < 8; ++j) {
      float v = pr[lane + 32 * j];
      vals[j] = v; sum += v; sq += v * v;
    }
#pragma unroll
    for (int d = 1; d < 32; d <<= 1) { sum += __shfl_xor(sum, d); sq += __shfl_xor(sq, d); }
    float mean = sum * (1.f / 256.f);
    float var  = sq * (1.f / 256.f) - mean * mean;
    float rs   = rsqrtf(var + 1e-5f);
#pragma unroll
    for (int j = 0; j < 8; ++j) {
      int c = lane + 32 * j;
      sXN[tok][c] = (_Float16)((vals[j] - mean) * rs * lng[c] + lnb[c]);
    }
  }
  __syncthreads();

  // Phase 2: QKV GEMM (16x768x256); wave covers n in [96w, 96w+96)
  {
    v8f zero = {};
    v8f acc[6];
#pragma unroll
    for (int f = 0; f < 6; ++f) acc[f] = zero;
    for (int kc = 0; kc < CH; kc += 32) {
      v16h a = ld_afrag(&sXN[0][kc], CH, lane);
#pragma unroll
      for (int f = 0; f < 6; ++f) {
        v16h bf = ld_bfrag_h(qkvw, CH, 96 * w + 16 * f, kc, lane);
        acc[f] = __builtin_amdgcn_wmma_f32_16x16x32_f16(false, a, false, bf, (short)0, acc[f], false, false);
      }
    }
#pragma unroll
    for (int f = 0; f < 6; ++f) {
      int n = 96 * w + 16 * f + nlo;
      float bsv = qkvb[n];
#pragma unroll
      for (int j = 0; j < 8; ++j) sQKV[j + 8 * hi][n] = (_Float16)(acc[f][j] + bsv);
    }
  }
  __syncthreads();

  // Phase 3: attention for head h = w (hd=32 == WMMA K exactly)
  {
    int h = w;
    v16h qa = ld_afrag(&sQKV[0][32 * h], 768, lane);   // A = q (16x32)
    v16h kb;                                           // B = k^T (32x16)
#pragma unroll
    for (int i = 0; i < 16; ++i) kb[i] = sQKV[nlo][256 + 32 * h + 16 * hi + i];
    v8f s = {};
    s = __builtin_amdgcn_wmma_f32_16x16x32_f16(false, qa, false, kb, (short)0, s, false, false);
    const float scale = 0.17677669529663689f;  // 1/sqrt(32)
    float pv[8];
#pragma unroll
    for (int j = 0; j < 8; ++j) {
      float x = s[j] * scale;
      float mx = x;
#pragma unroll
      for (int d = 1; d < 16; d <<= 1) mx = fmaxf(mx, __shfl_xor(mx, d));
      float e = __expf(x - mx);
      float sm = e;
#pragma unroll
      for (int d = 1; d < 16; d <<= 1) sm += __shfl_xor(sm, d);
      pv[j] = e / sm;
    }
#pragma unroll
    for (int j = 0; j < 8; ++j) sP[w][j + 8 * hi][nlo] = (_Float16)pv[j];
    // A = P padded to 16x32 (K>=16 -> 0)
    v16h pa;
#pragma unroll
    for (int i = 0; i < 16; ++i) {
      int k = 16 * hi + i;
      pa[i] = (k < 16) ? sP[w][nlo][k] : (_Float16)0.f;
    }
#pragma unroll
    for (int nf = 0; nf < 2; ++nf) {
      v16h vb;  // B = v (K=token rows 0..15 valid, N = d-half)
#pragma unroll
      for (int i = 0; i < 16; ++i)
        vb[i] = hi ? (_Float16)0.f : sQKV[i][512 + 32 * h + 16 * nf + nlo];
      v8f o = {};
      o = __builtin_amdgcn_wmma_f32_16x16x32_f16(false, pa, false, vb, (short)0, o, false, false);
#pragma unroll
      for (int j = 0; j < 8; ++j) sAO[j + 8 * hi][32 * h + 16 * nf + nlo] = (_Float16)o[j];
    }
  }
  __syncthreads();

  // Phase 4: proj GEMM (16x256x256); wave covers n in [32w, 32w+32); scatter to pixels
  {
    v8f pc0 = {}, pc1 = {};
    for (int kc = 0; kc < CH; kc += 32) {
      v16h a  = ld_afrag(&sAO[0][kc], CH, lane);
      v16h b0 = ld_bfrag_h(projw, CH, 32 * w, kc, lane);
      v16h b1 = ld_bfrag_h(projw, CH, 32 * w + 16, kc, lane);
      pc0 = __builtin_amdgcn_wmma_f32_16x16x32_f16(false, a, false, b0, (short)0, pc0, false, false);
      pc1 = __builtin_amdgcn_wmma_f32_16x16x32_f16(false, a, false, b1, (short)0, pc1, false, false);
    }
    float bs0 = projb[32 * w + nlo], bs1 = projb[32 * w + 16 + nlo];
#pragma unroll
    for (int j = 0; j < 8; ++j) {
      int tok = j + 8 * hi;
      int row = hb * 4 + (tok >> 2), col = wb * 4 + (tok & 3);
      long m = (long)b * HW + row * WIM + col;
      outp[m * CH + 32 * w + nlo]      = pc0[j] + bs0;
      outp[m * CH + 32 * w + 16 + nlo] = pc1[j] + bs1;
    }
  }
}

// ---- GroupNorm stats over buf[M][256], per (batch, group of 32 ch) ----
__global__ __launch_bounds__(256) void k_gnstats(const float* __restrict__ buf,
                                                 float* __restrict__ mu, float* __restrict__ rsg) {
  int bg = blockIdx.x;  // 0..31
  int b = bg >> 3, g = bg & 7;
  int t = threadIdx.x, cc = t & 31, hh = t >> 5;
  float s = 0.f, q = 0.f;
  const float* base = buf + (long)b * HW * CH + g * 32 + cc;
  for (int hw = hh; hw < HW; hw += 8) {
    float v = base[(long)hw * CH];
    s += v; q += v * v;
  }
#pragma unroll
  for (int d = 1; d < 32; d <<= 1) { s += __shfl_xor(s, d); q += __shfl_xor(q, d); }
  __shared__ float ss_[8], sq_[8];
  int w = t >> 5, lane = t & 31;
  if (lane == 0) { ss_[w] = s; sq_[w] = q; }
  __syncthreads();
  if (t == 0) {
    float S = 0.f, Q = 0.f;
    for (int i = 0; i < 8; ++i) { S += ss_[i]; Q += sq_[i]; }
    const float inv = 1.f / (32.f * (float)HW);
    float m = S * inv;
    float var = Q * inv - m * m;
    mu[bg]  = m;
    rsg[bg] = rsqrtf(var + 1e-5f);
  }
}

// ---- LSTM pointwise kernels ----
__global__ void k_ig(float* __restrict__ ipart, const float* __restrict__ ggate, int n) {
  int idx = blockIdx.x * 256 + threadIdx.x;
  if (idx < n) ipart[idx] = sigf(ipart[idx]) * tanhf(ggate[idx]);
}

__global__ void k_cnext(const float* __restrict__ fgate, const float* __restrict__ cin,
                        const float* __restrict__ ig, float* __restrict__ cout, int n) {
  long idx = (long)blockIdx.x * 256 + threadIdx.x;  // (b,c,hw) layout
  if (idx >= n) return;
  long b = idx >> 22;
  long rem = idx & 4194303;
  int c = (int)(rem >> 14), hw = (int)(rem & 16383);
  long m = b * HW + hw;
  cout[idx] = sigf(fgate[m * CH + c]) * cin[idx] + ig[m * CH + c];
}

__global__ void k_tgate(const float* __restrict__ ogate, const float* __restrict__ cnext,
                        float* __restrict__ tout, int n) {
  long idx = (long)blockIdx.x * 256 + threadIdx.x;  // [m][c] layout
  if (idx >= n) return;
  long m = idx >> 8; int c = (int)(idx & 255);
  long b = m >> 14; long hw = m & 16383;
  long cidx = ((b * CH + c) << 14) + hw;
  tout[idx] = sigf(ogate[idx]) * tanhf(cnext[cidx]);
}

__global__ void k_gnapply(const float* __restrict__ tbuf, const float* __restrict__ mu,
                          const float* __restrict__ rsg, const float* __restrict__ gng,
                          const float* __restrict__ gnb, float* __restrict__ hout, int n) {
  long idx = (long)blockIdx.x * 256 + threadIdx.x;  // (b,c,hw)
  if (idx >= n) return;
  long b = idx >> 22;
  long rem = idx & 4194303;
  int c = (int)(rem >> 14), hw = (int)(rem & 16383);
  int g = c >> 5;
  long m = b * HW + hw;
  float v = tbuf[m * CH + c];
  hout[idx] = (v - mu[b * 8 + g]) * rsg[b * 8 + g] * gng[c] + gnb[c];
}

extern "C" void kernel_launch(void* const* d_in, const int* in_sizes, int n_in,
                              void* d_out, int out_size, void* d_ws, size_t ws_size,
                              hipStream_t stream) {
  const float* x    = (const float*)d_in[0];
  const float* hN   = (const float*)d_in[1];
  const float* cN   = (const float*)d_in[2];
  const float* ipw  = (const float*)d_in[3];
  const float* ipb  = (const float*)d_in[4];
  const float* lng  = (const float*)d_in[5];
  const float* lnb  = (const float*)d_in[6];
  const float* qkvw = (const float*)d_in[7];
  const float* qkvb = (const float*)d_in[8];
  const float* pw   = (const float*)d_in[9];
  const float* pb   = (const float*)d_in[10];
  const float* gw   = (const float*)d_in[11];
  const float* gb   = (const float*)d_in[12];
  const float* gng  = (const float*)d_in[13];
  const float* gnb  = (const float*)d_in[14];

  const long PLANE = (long)MTOT * CH;  // 16,777,216 elements (64 MB)
  float* R0 = (float*)d_ws;            // p (in_proj out), later t = o*tanh(cnext)
  float* R1 = R0 + PLANE;              // attention/proj output (pre-GN)
  float* R2 = R1 + PLANE;              // gate i  -> i*g partial
  float* R3 = R2 + PLANE;              // gate g, then f, then o
  float* ST = R3 + PLANE;              // stats (128 floats)
  float *mu1 = ST, *rs1 = ST + 32, *mu2 = ST + 64, *rs2 = ST + 96;

  // f16 weight copies (pre-converted once per launch)
  const int N_IPW = 256 * 320, N_QKV = 768 * 256, N_PRJ = 256 * 256, N_GTS = 1024 * 256;
  _Float16* wf_ip   = (_Float16*)(ST + 128);
  _Float16* wf_qkv  = wf_ip + N_IPW;
  _Float16* wf_proj = wf_qkv + N_QKV;
  _Float16* wf_gts  = wf_proj + N_PRJ;

  float* hout = (float*)d_out;
  float* cout = hout + PLANE;

  dim3 blk(256);
  dim3 gG(MTOT / 64, 4);
  int n = (int)PLANE;
  int nb = (n + 255) / 256;

  // 0. pre-convert weights fp32 -> f16 (tiny; halves weight fetch bandwidth in GEMMs)
  k_cvt<<<(N_IPW + 255) / 256, blk, 0, stream>>>(ipw, wf_ip, N_IPW);
  k_cvt<<<(N_QKV + 255) / 256, blk, 0, stream>>>(qkvw, wf_qkv, N_QKV);
  k_cvt<<<(N_PRJ + 255) / 256, blk, 0, stream>>>(pw, wf_proj, N_PRJ);
  k_cvt<<<(N_GTS + 255) / 256, blk, 0, stream>>>(gw, wf_gts, N_GTS);

  // 1. in_proj: p = concat(x,h) @ W^T + b
  k_gemm<0><<<gG, blk, 0, stream>>>(x, hN, wf_ip, ipb, nullptr, nullptr, nullptr, nullptr, R0, 320);
  // 2. fused LN + QKV + windowed attention + proj
  k_attn<<<4096, blk, 0, stream>>>(R0, lng, lnb, wf_qkv, qkvb, wf_proj, pb, R1);
  // 3. GroupNorm #1 stats
  k_gnstats<<<32, blk, 0, stream>>>(R1, mu1, rs1);
  // 4. gates (GN fused into A-load), split per gate: i, g, f, o
  k_gemm<1><<<gG, blk, 0, stream>>>(R1, nullptr, wf_gts + 0L * CH * CH, gb + 0 * CH, mu1, rs1, gng, gnb, R2, CH);
  k_gemm<1><<<gG, blk, 0, stream>>>(R1, nullptr, wf_gts + 3L * CH * CH, gb + 3 * CH, mu1, rs1, gng, gnb, R3, CH);
  k_ig<<<nb, blk, 0, stream>>>(R2, R3, n);  // R2 = sigmoid(i)*tanh(g)
  k_gemm<1><<<gG, blk, 0, stream>>>(R1, nullptr, wf_gts + 1L * CH * CH, gb + 1 * CH, mu1, rs1, gng, gnb, R3, CH);
  k_cnext<<<nb, blk, 0, stream>>>(R3, cN, R2, cout, n);  // cnext -> d_out (second output)
  k_gemm<1><<<gG, blk, 0, stream>>>(R1, nullptr, wf_gts + 2L * CH * CH, gb + 2 * CH, mu1, rs1, gng, gnb, R3, CH);
  k_tgate<<<nb, blk, 0, stream>>>(R3, cout, R0, n);      // R0 = o*tanh(cnext)
  // 5. GroupNorm #2 -> hnext
  k_gnstats<<<32, blk, 0, stream>>>(R0, mu2, rs2);
  k_gnapply<<<nb, blk, 0, stream>>>(R0, mu2, rs2, gng, gnb, hout, n);
}
